// SSDBBox_61624190763481
// MI455X (gfx1250) — compile-verified, matrix operand order checked by reading, not run on hardware
//
#include <hip/hip_runtime.h>
#include <hip/hip_bf16.h>

typedef __attribute__((ext_vector_type(16))) _Float16 v16h;
typedef __attribute__((ext_vector_type(8)))  float    v8f;

#define NUM_CLASSES 80
#define C_OUT       81
#define EXS         96          // padded row stride (3 x K=32 WMMA chunks)
#define SCORE_THR   0.02f
#define IOU_THR     0.45f
#define IMG_MAXF    320.0f
#define MAX_RATIO   4.1351666f
#define K_TOTAL     5480
#define K_PAD       8192
#define NBINS       2048
#define BATCH       4

__constant__ int   c_FEAT[6]   = {20, 10, 5, 3, 2, 1};
__constant__ float c_STRIDE[6] = {16.f, 32.f, 64.f, 107.f, 160.f, 320.f};
__constant__ float c_MINS[6]   = {48.f, 100.f, 150.f, 202.f, 253.f, 304.f};
__constant__ float c_MAXS[6]   = {100.f, 150.f, 202.f, 253.f, 304.f, 320.f};
__constant__ int   c_KLVL[6]   = {1000, 1000, 1000, 1000, 1000, 480};
__constant__ int   c_LBASE[6]  = {0, 1000, 2000, 3000, 4000, 5000};

struct InPtrs { const float* cls[6]; const float* bb[6]; };

// ---------------------------------------------------------------------------
// helpers
// ---------------------------------------------------------------------------
__device__ __forceinline__ void get_prior(int lvl, int a, int s, int f, float* p) {
    int x = s % f; int y = s / f;
    float base = c_MINS[lvl];
    float scale = 1.f, ratio = 1.f;
    if (a == 1) scale = sqrtf(c_MAXS[lvl] / c_MINS[lvl]);
    else if (a == 2) ratio = 0.5f;
    else if (a == 3) ratio = 2.0f;
    else if (a == 4) ratio = 1.0f / 3.0f;
    else if (a == 5) ratio = 3.0f;
    float hr = sqrtf(ratio), wr = 1.0f / hr;
    float w = base * scale * wr, h = base * scale * hr;
    float st = c_STRIDE[lvl];
    float cx = st * 0.5f + (float)x * st;
    float cy = st * 0.5f + (float)y * st;
    p[0] = cx - 0.5f * w; p[1] = cy - 0.5f * h;
    p[2] = cx + 0.5f * w; p[3] = cy + 0.5f * h;
}

__device__ __forceinline__ void decode_box(const float* pr, const float* dl, float* box) {
    float dx = dl[0] * 0.1f, dy = dl[1] * 0.1f;
    float dw = dl[2] * 0.2f, dh = dl[3] * 0.2f;
    dw = fminf(fmaxf(dw, -MAX_RATIO), MAX_RATIO);
    dh = fminf(fmaxf(dh, -MAX_RATIO), MAX_RATIO);
    float pcx = (pr[0] + pr[2]) * 0.5f, pcy = (pr[1] + pr[3]) * 0.5f;
    float pw = pr[2] - pr[0], ph = pr[3] - pr[1];
    float gx = pcx + pw * dx, gy = pcy + ph * dy;
    float gw = pw * __expf(dw), gh = ph * __expf(dh);
    box[0] = fminf(fmaxf(gx - 0.5f * gw, 0.f), IMG_MAXF);
    box[1] = fminf(fmaxf(gy - 0.5f * gh, 0.f), IMG_MAXF);
    box[2] = fminf(fmaxf(gx + 0.5f * gw, 0.f), IMG_MAXF);
    box[3] = fminf(fmaxf(gy + 0.5f * gh, 0.f), IMG_MAXF);
}

// Softmax over 81 classes for a tile of 16 consecutive spatial cells of one
// anchor.  The denominator sum_c exp(x_c - max) is computed as a chain of 3
// v_wmma_f32_16x16x32_f16 against an all-ones B matrix (K padded to 96 with
// zeros, so the fragment fill is an unconditional contiguous LDS read).
// Tiling by (anchor, 16 spatial) makes the global loads coalesced:
// cls[(a*81+c)*f*f + s0 .. s0+15] is contiguous per class.
__device__ void softmax_tile_wmma(const float* cls, int a, int s0, int ff,
                                  int lane, float* ex /*16*EXS*/,
                                  float* rmax /*16*/, float* den /*16*/) {
    // phase 1: cooperative coalesced load of raw logits into LDS
    for (int e = lane; e < 16 * C_OUT; e += 32) {
        int c = e >> 4, m = e & 15;       // half-wave reads 16 contiguous cells
        int s = s0 + m;
        float v = (s < ff) ? cls[(size_t)(a * C_OUT + c) * ff + s] : 0.f;
        ex[m * EXS + c] = v;
    }
    // zero the K-padding columns 81..95 (makes WMMA fill unconditional)
    for (int e = lane; e < 16 * (EXS - C_OUT); e += 32) {
        int m = e / (EXS - C_OUT), c = C_OUT + e % (EXS - C_OUT);
        ex[m * EXS + c] = 0.f;
    }
    __builtin_amdgcn_wave_barrier();

    // phase 2: per-row max (lanes 0..15, reads from LDS)
    if (lane < 16) {
        float mx = -1e30f;
        for (int c = 0; c < C_OUT; ++c) mx = fmaxf(mx, ex[lane * EXS + c]);
        rmax[lane] = mx;
    }
    __builtin_amdgcn_wave_barrier();

    // phase 3: cooperative in-place exponentiation (invalid rows -> 0)
    for (int e = lane; e < 16 * C_OUT; e += 32) {
        int c = e >> 4, m = e & 15;
        float v = ex[m * EXS + c];
        bool vr = (s0 + m) < ff;
        ex[m * EXS + c] = vr ? __expf(v - rmax[m]) : 0.f;
    }
    __builtin_amdgcn_wave_barrier();

    // phase 4: WMMA row-sum.  A-layout: lanes L and L+16 both own row M=L&15,
    // each lane supplies 16 distinct K positions of its row (K order within a
    // row is irrelevant because B is all-ones).
    v16h ones;
    for (int i = 0; i < 16; ++i) ones[i] = (_Float16)1.0f;
    v8f acc = {};
    int m  = lane & 15;
    int hi = (lane >> 4) & 1;
    for (int k0 = 0; k0 < EXS; k0 += 32) {
        const float* rp = ex + m * EXS + k0 + hi * 16;
        v16h afrag;
        for (int i = 0; i < 16; ++i) afrag[i] = (_Float16)rp[i];  // 16 contiguous f32
        acc = __builtin_amdgcn_wmma_f32_16x16x32_f16(
            false, afrag, false, ones, (short)0, acc, false, false);
    }
    // D layout: lane 0 holds D[0..7][0] (rows 0..7), lane 16 holds rows 8..15.
    if (lane == 0)  for (int r = 0; r < 8; ++r) den[r]     = acc[r];
    if (lane == 16) for (int r = 0; r < 8; ++r) den[8 + r] = acc[r];
    __builtin_amdgcn_wave_barrier();
}

// ---------------------------------------------------------------------------
// kernels
// ---------------------------------------------------------------------------
__global__ void init_kernel(int* p, int n) {
    int i = blockIdx.x * blockDim.x + threadIdx.x;
    if (i < n) p[i] = 0;
}

// Pass A: per (img,level) block.  Histogram of score bits -> exact top-k
// cutoff bin + tie quota (radix select instead of a 192K-element sort).
__global__ __launch_bounds__(256)
void passA_kernel(InPtrs in, int* cutBin, int* quota) {
    __shared__ float sh_ex[8][16 * EXS];
    __shared__ float sh_rmax[8][16];
    __shared__ float sh_den[8][16];
    __shared__ int   sh_hist[NBINS];
    int img = blockIdx.x / 6, lvl = blockIdx.x % 6;
    int f = c_FEAT[lvl];
    int ff = f * f;
    const float* cls = in.cls[lvl] + (size_t)img * 486 * ff;
    __builtin_prefetch(cls, 0, 1);              // global_prefetch_b8
    __builtin_amdgcn_s_wait_tensorcnt(0);       // gfx1250 split-counter path
    int wave = threadIdx.x >> 5, lane = threadIdx.x & 31;
    for (int i = threadIdx.x; i < NBINS; i += blockDim.x) sh_hist[i] = 0;
    __syncthreads();

    int stiles = (ff + 15) / 16;
    int tiles = 6 * stiles;
    for (int t = wave; t < tiles; t += 8) {
        int a = t % 6, s0 = (t / 6) * 16;
        float* ex = sh_ex[wave];
        softmax_tile_wmma(cls, a, s0, ff, lane, ex, sh_rmax[wave], sh_den[wave]);
        const float* den = sh_den[wave];
        for (int e = lane; e < 16 * NUM_CLASSES; e += 32) {
            int m = e / NUM_CLASSES, c = e % NUM_CLASSES;
            if (s0 + m < ff) {
                float p = ex[m * EXS + c] / den[m];
                if (p > SCORE_THR) {
                    int b = (int)((__float_as_uint(p) - 0x3C000000u) >> 15);
                    if (b > NBINS - 1) b = NBINS - 1;
                    atomicAdd(&sh_hist[b], 1);
                }
            }
        }
    }
    __syncthreads();
    if (threadIdx.x == 0) {
        int k = c_KLVL[lvl];
        int cum = 0, T = -1, q = 0;
        for (int b = NBINS - 1; b >= 0; --b) {
            int h = sh_hist[b];
            if (cum + h >= k) { T = b; q = k - cum; break; }
            cum += h;
        }
        cutBin[blockIdx.x] = T;   // -1 => fewer than k candidates: take all
        quota[blockIdx.x]  = q;
    }
}

// Pass B: recompute scores (bit-identical), append accepted candidates with
// decoded boxes into per-image per-level slots.
__global__ __launch_bounds__(256)
void passB_kernel(InPtrs in, const int* cutBin, const int* quota,
                  int* levelCount, int* tieCount,
                  float* cand_score, float* cand_box, int* cand_label) {
    __shared__ float sh_ex[8][16 * EXS];
    __shared__ float sh_rmax[8][16];
    __shared__ float sh_den[8][16];
    int img = blockIdx.x / 6, lvl = blockIdx.x % 6;
    int f = c_FEAT[lvl];
    int ff = f * f;
    const float* cls = in.cls[lvl] + (size_t)img * 486 * ff;
    const float* bb  = in.bb[lvl]  + (size_t)img * 24 * ff;
    __builtin_prefetch(bb, 0, 1);
    int wave = threadIdx.x >> 5, lane = threadIdx.x & 31;
    int T = cutBin[blockIdx.x], q = quota[blockIdx.x];
    int base = c_LBASE[lvl];
    int klvl = c_KLVL[lvl];

    int stiles = (ff + 15) / 16;
    int tiles = 6 * stiles;
    for (int t = wave; t < tiles; t += 8) {
        int a = t % 6, s0 = (t / 6) * 16;
        float* ex = sh_ex[wave];
        softmax_tile_wmma(cls, a, s0, ff, lane, ex, sh_rmax[wave], sh_den[wave]);
        const float* den = sh_den[wave];
        for (int e = lane; e < 16 * NUM_CLASSES; e += 32) {
            int m = e / NUM_CLASSES, c = e % NUM_CLASSES;
            int s = s0 + m;
            if (s >= ff) continue;
            float p = ex[m * EXS + c] / den[m];
            if (p <= SCORE_THR) continue;
            int b = (int)((__float_as_uint(p) - 0x3C000000u) >> 15);
            if (b > NBINS - 1) b = NBINS - 1;
            bool take = (b > T);
            if (!take && b == T)
                take = atomicAdd(&tieCount[blockIdx.x], 1) < q;
            if (!take) continue;
            int pos = atomicAdd(&levelCount[blockIdx.x], 1);
            if (pos >= klvl) continue;    // safety cap
            int slot = img * K_TOTAL + base + pos;

            float prior[4]; get_prior(lvl, a, s, f, prior);
            float dl[4];
            for (int j = 0; j < 4; ++j)
                dl[j] = bb[(size_t)(a * 4 + j) * ff + s];
            float box[4]; decode_box(prior, dl, box);
            cand_score[slot] = p;
            cand_label[slot] = c;
            for (int j = 0; j < 4; ++j) cand_box[slot * 4 + j] = box[j];
        }
    }
}

// Per-image descending bitonic sort of 8192 (score,slot) keys in LDS, then
// gather sorted candidate data into contiguous arrays.
__global__ __launch_bounds__(1024)
void sort_kernel(const float* cand_score, const float* cand_box, const int* cand_label,
                 float* sscore, float* sbox, int* slabel) {
    __shared__ unsigned int   kk[K_PAD];   // score bits (positive -> monotone)
    __shared__ unsigned short vv[K_PAD];   // slot payload
    int img = blockIdx.x;
    const float* cs = cand_score + (size_t)img * K_TOTAL;

    for (int i = threadIdx.x; i < K_PAD; i += blockDim.x) {
        unsigned int sb = 0u; unsigned short slot = 0xFFFFu;
        if (i < K_TOTAL) { sb = __float_as_uint(cs[i]); slot = (unsigned short)i; }
        kk[i] = sb; vv[i] = slot;
    }
    __syncthreads();

    for (int k = 2; k <= K_PAD; k <<= 1) {
        for (int j = k >> 1; j > 0; j >>= 1) {
            for (int i = threadIdx.x; i < K_PAD; i += blockDim.x) {
                int l = i ^ j;
                if (l > i) {
                    bool desc = ((i & k) == 0);
                    unsigned int a = kk[i], b = kk[l];
                    bool sw = desc ? (a < b) : (a > b);
                    if (sw) {
                        kk[i] = b; kk[l] = a;
                        unsigned short ta = vv[i]; vv[i] = vv[l]; vv[l] = ta;
                    }
                }
            }
            __syncthreads();
        }
    }

    const float* cbx = cand_box   + (size_t)img * K_TOTAL * 4;
    const int*   clb = cand_label + (size_t)img * K_TOTAL;
    float* os = sscore + (size_t)img * K_TOTAL;
    float* ob = sbox   + (size_t)img * K_TOTAL * 4;
    int*   ol = slabel + (size_t)img * K_TOTAL;
    for (int i = threadIdx.x; i < K_TOTAL; i += blockDim.x) {
        int slot = vv[i];
        if (slot < K_TOTAL) {
            os[i] = __uint_as_float(kk[i]);
            ol[i] = clb[slot];
            for (int j = 0; j < 4; ++j) ob[i * 4 + j] = cbx[slot * 4 + j];
        } else {
            os[i] = 0.f; ol[i] = 0;
            for (int j = 0; j < 4; ++j) ob[i * 4 + j] = 0.f;
        }
    }
}

// Greedy class-aware NMS (label offset trick) + final output write.
__global__ __launch_bounds__(1024)
void nms_kernel(const float* sbox, const float* sscore, const int* slabel,
                float* out) {
    __shared__ unsigned char alive[K_TOTAL];
    int img = blockIdx.x;
    const float* bx = sbox   + (size_t)img * K_TOTAL * 4;
    const float* sc = sscore + (size_t)img * K_TOTAL;
    const int*   lb = slabel + (size_t)img * K_TOTAL;

    for (int i = threadIdx.x; i < K_TOTAL; i += blockDim.x)
        alive[i] = (sc[i] > SCORE_THR) ? 1 : 0;
    __syncthreads();

    for (int i = 0; i < K_TOTAL; ++i) {
        if (alive[i]) {
            float off_i = (float)lb[i] * (IMG_MAXF + 1.0f);
            float ax0 = bx[4 * i + 0] + off_i, ay0 = bx[4 * i + 1] + off_i;
            float ax1 = bx[4 * i + 2] + off_i, ay1 = bx[4 * i + 3] + off_i;
            float ai = (ax1 - ax0) * (ay1 - ay0);
            for (int j = i + 1 + threadIdx.x; j < K_TOTAL; j += blockDim.x) {
                if (!alive[j]) continue;
                float off_j = (float)lb[j] * (IMG_MAXF + 1.0f);
                float bx0 = bx[4 * j + 0] + off_j, by0 = bx[4 * j + 1] + off_j;
                float bx1 = bx[4 * j + 2] + off_j, by1 = bx[4 * j + 3] + off_j;
                float aj = (bx1 - bx0) * (by1 - by0);
                float iw = fmaxf(fminf(ax1, bx1) - fmaxf(ax0, bx0), 0.f);
                float ih = fmaxf(fminf(ay1, by1) - fmaxf(ay0, by0), 0.f);
                float inter = iw * ih;
                float iou = inter / (ai + aj - inter);
                if (iou > IOU_THR) alive[j] = 0;
            }
        }
        __syncthreads();
    }

    // outputs: dets [B,K,5], labels [B,K], keep [B,K]  (all as f32, flat)
    float* dets  = out;
    float* lbls  = out + (size_t)BATCH * K_TOTAL * 5;
    float* keepo = lbls + (size_t)BATCH * K_TOTAL;
    for (int i = threadIdx.x; i < K_TOTAL; i += blockDim.x) {
        float kf = alive[i] ? 1.f : 0.f;
        size_t db = (size_t)img * K_TOTAL * 5 + (size_t)i * 5;
        for (int j = 0; j < 4; ++j) dets[db + j] = bx[4 * i + j] * kf;
        dets[db + 4] = sc[i] * kf;
        lbls[(size_t)img * K_TOTAL + i]  = (float)lb[i];
        keepo[(size_t)img * K_TOTAL + i] = kf;
    }
}

// ---------------------------------------------------------------------------
// launcher
// ---------------------------------------------------------------------------
extern "C" void kernel_launch(void* const* d_in, const int* in_sizes, int n_in,
                              void* d_out, int out_size, void* d_ws, size_t ws_size,
                              hipStream_t stream) {
    (void)in_sizes; (void)n_in; (void)out_size; (void)ws_size;
    InPtrs in;
    for (int i = 0; i < 6; ++i) {
        in.cls[i] = (const float*)d_in[i];
        in.bb[i]  = (const float*)d_in[6 + i];
    }
    int*   wsi = (int*)d_ws;
    float* wsf = (float*)d_ws;
    // ws layout (4-byte units):
    int*   levelCount = wsi;            // 24
    int*   tieCount   = wsi + 24;       // 24
    int*   cutBin     = wsi + 48;       // 24
    int*   quota      = wsi + 72;       // 24
    float* cand_score = wsf + 96;       // 4*5480
    int*   cand_label = wsi + 22016;    // 4*5480
    float* cand_box   = wsf + 43936;    // 4*5480*4
    float* sbox       = wsf + 131616;   // 4*5480*4
    float* sscore     = wsf + 219296;   // 4*5480
    int*   slabel     = wsi + 241216;   // 4*5480
    const int zeroN = 131616;           // counters + candidate arrays

    init_kernel<<<(zeroN + 255) / 256, 256, 0, stream>>>(wsi, zeroN);
    passA_kernel<<<BATCH * 6, 256, 0, stream>>>(in, cutBin, quota);
    passB_kernel<<<BATCH * 6, 256, 0, stream>>>(in, cutBin, quota,
                                                levelCount, tieCount,
                                                cand_score, cand_box, cand_label);
    sort_kernel<<<BATCH, 1024, 0, stream>>>(cand_score, cand_box, cand_label,
                                            sscore, sbox, slabel);
    nms_kernel<<<BATCH, 1024, 0, stream>>>(sbox, sscore, slabel, (float*)d_out);
}
